// GATConvOp_61976378081384
// MI455X (gfx1250) — matrix operand order repaired and checked by reading, not compiled, not run on hardware
//
#include <hip/hip_runtime.h>
#include <hip/hip_bf16.h>

typedef __attribute__((ext_vector_type(16))) _Float16 v16h;
typedef __attribute__((ext_vector_type(8)))  float    v8f;

#define GAT_N 20000
#define GAT_E 320000
#define GAT_F 128
#define NEG_SLOPE 0.2f

// ---------------------------------------------------------------------------
// Fragment packing: convert f32 matrices into WMMA f16 fragment order so the
// GEMM inner loop is pure contiguous v16h loads (coalesced b128 pairs).
//
// A fragment (16x32, ISA 7.12.2 "16-bit A-Matrix 16x32"):
//   lane<16 : row=lane,    halves i -> K = grp*16 +  win*2+lo        (0..7,16..23)
//   lane>=16: row=lane-16, halves i -> K = grp*16 + 8 + win*2+lo     (8..15,24..31)
// B fragment (32x16): col = lane&15, half i -> K = (i>=8?16:0)+(lane>=16?8:0)+(i&7)
// Packed layout: frag[(tile_m_or_n * Ktiles + kt) * 32 + lane] = v16h
// ---------------------------------------------------------------------------
__global__ __launch_bounds__(32)
void pack_a_frag(const float* __restrict__ A, _Float16* __restrict__ Ap,
                 int K, int Ktiles) {
    const int tm = blockIdx.x, tk = blockIdx.y;
    const int lane = threadIdx.x & 31;
    const bool hi  = lane >= 16;
    const int r16  = lane & 15;
    v16h v;
#pragma unroll
    for (int i = 0; i < 16; ++i) {
        const int vg  = i >> 1, lo = i & 1;
        const int grp = vg >> 2, win = vg & 3;
        const int ka  = grp * 16 + (hi ? 8 : 0) + win * 2 + lo;
        v[i] = (_Float16)A[(size_t)(tm * 16 + r16) * K + tk * 32 + ka];
    }
    ((v16h*)Ap)[((size_t)tm * Ktiles + tk) * 32 + lane] = v;
}

__global__ __launch_bounds__(32)
void pack_b_frag(const float* __restrict__ B, _Float16* __restrict__ Bp,
                 int Nn, int Ktiles) {
    const int tn = blockIdx.x, tk = blockIdx.y;
    const int lane = threadIdx.x & 31;
    const bool hi  = lane >= 16;
    const int r16  = lane & 15;
    v16h v;
#pragma unroll
    for (int i = 0; i < 16; ++i) {
        const int kb = ((i & 8) ? 16 : 0) + (hi ? 8 : 0) + (i & 7);
        v[i] = (_Float16)B[(size_t)(tk * 32 + kb) * Nn + tn * 16 + r16];
    }
    ((v16h*)Bp)[((size_t)tn * Ktiles + tk) * 32 + lane] = v;
}

// ---------------------------------------------------------------------------
// Packed-fragment WMMA GEMM. 256 threads = 8 waves per block; block covers a
// 16-row strip; wave w owns NT consecutive 16-col tiles (A reused NT times).
// EXEC all-1s everywhere (no divergence) as V_WMMA requires.
// ---------------------------------------------------------------------------
template <int NT>
__global__ __launch_bounds__(256)
void gemm_wmma_packed(const _Float16* __restrict__ Ap, const _Float16* __restrict__ Bp,
                      float* __restrict__ C, int Ktiles, int Nn) {
    const int wave = threadIdx.x >> 5;
    const int lane = threadIdx.x & 31;
    const int tileM  = blockIdx.x;
    const int tileN0 = wave * NT;
    const bool hi = lane >= 16;
    const int r16 = lane & 15;

    const v16h* Av = (const v16h*)Ap;
    const v16h* Bv = (const v16h*)Bp;

    v8f acc[NT];
#pragma unroll
    for (int t = 0; t < NT; ++t) acc[t] = (v8f){};

    for (int kt = 0; kt < Ktiles; ++kt) {
        const v16h av = Av[((size_t)tileM * Ktiles + kt) * 32 + lane];
#pragma unroll
        for (int t = 0; t < NT; ++t) {
            const v16h bv = Bv[((size_t)(tileN0 + t) * Ktiles + kt) * 32 + lane];
            acc[t] = __builtin_amdgcn_wmma_f32_16x16x32_f16(
                false, av, false, bv, (short)0, acc[t], false, false);
        }
    }
#pragma unroll
    for (int t = 0; t < NT; ++t) {
        const int col = (tileN0 + t) * 16 + r16;
#pragma unroll
        for (int r = 0; r < 8; ++r)
            C[(size_t)(tileM * 16 + r + (hi ? 8 : 0)) * Nn + col] = acc[t][r];
    }
}

// ---------------------------------------------------------------------------
// el[n,h] = dot(h[n,h,:], attn_l[h,:]); er likewise. One wave per (node,head).
// ---------------------------------------------------------------------------
__global__ void attn_dots(const float* __restrict__ h, const float* __restrict__ al,
                          const float* __restrict__ ar, float* __restrict__ el,
                          float* __restrict__ er, int n, int heads) {
    const int wid  = (int)((blockIdx.x * (size_t)blockDim.x + threadIdx.x) >> 5);
    const int lane = threadIdx.x & 31;
    if (wid >= n * heads) return;
    const int node = wid / heads, hh = wid % heads;
    const float* hp  = h  + ((size_t)node * heads + hh) * GAT_F;
    const float* alp = al + (size_t)hh * GAT_F;
    const float* arp = ar + (size_t)hh * GAT_F;
    float sl = 0.f, sr = 0.f;
#pragma unroll
    for (int i = 0; i < GAT_F / 32; ++i) {
        const float v = hp[lane + i * 32];
        sl += v * alp[lane + i * 32];
        sr += v * arp[lane + i * 32];
    }
#pragma unroll
    for (int off = 16; off > 0; off >>= 1) {        // wave32 reduction
        sl += __shfl_down(sl, off, 32);
        sr += __shfl_down(sr, off, 32);
    }
    if (lane == 0) { el[wid] = sl; er[wid] = sr; }
}

// ---------------------------------------------------------------------------
// helpers
// ---------------------------------------------------------------------------
__device__ __forceinline__ void atomicMaxF32(float* addr, float val) {
    unsigned int* ua = (unsigned int*)addr;
    unsigned int old = *ua;
    while (true) {
        const float f = __uint_as_float(old);
        if (f >= val) break;
        const unsigned int assumed = old;
        old = atomicCAS(ua, assumed, __float_as_uint(val));
        if (old == assumed) break;
    }
}

__global__ void fill_f32(float* __restrict__ p, float v, int n) {
    const int t = blockIdx.x * blockDim.x + threadIdx.x;
    if (t < n) p[t] = v;
}

__global__ void init_bias(float* __restrict__ o, const float* __restrict__ bias,
                          int n, int d) {
    const int t = blockIdx.x * blockDim.x + threadIdx.x;
    if (t < n * d) o[t] = bias[t % d];
}

// e = leaky_relu(el[src]+er[dst]); running segment-max over dst
__global__ void edge_score(const int* __restrict__ src, const int* __restrict__ dst,
                           const float* __restrict__ el, const float* __restrict__ er,
                           float* __restrict__ esc, float* __restrict__ emax,
                           int ne, int heads) {
    const int t = blockIdx.x * blockDim.x + threadIdx.x;
    if (t >= ne * heads) return;
    const int e = t / heads, hh = t % heads;
    float s = el[src[e] * heads + hh] + er[dst[e] * heads + hh];
    s = s > 0.f ? s : NEG_SLOPE * s;
    esc[t] = s;
    atomicMaxF32(&emax[dst[e] * heads + hh], s);
}

// a = exp(e - emax[dst]); denom[dst] += a
__global__ void edge_exp(const int* __restrict__ dst, float* __restrict__ esc,
                         const float* __restrict__ emax, float* __restrict__ denom,
                         int ne, int heads) {
    const int t = blockIdx.x * blockDim.x + threadIdx.x;
    if (t >= ne * heads) return;
    const int e = t / heads, hh = t % heads;
    const float a = __expf(esc[t] - emax[dst[e] * heads + hh]);
    esc[t] = a;
    atomicAdd(&denom[dst[e] * heads + hh], a);
}

// rst[dst] += (a/denom[dst]) * h[src] ; one wave per edge, float4 per lane
__global__ void edge_aggregate(const int* __restrict__ src, const int* __restrict__ dst,
                               const float* __restrict__ esc, const float* __restrict__ denom,
                               const float* __restrict__ h, float* __restrict__ rst,
                               int ne, int heads) {
    const int wid  = (int)((blockIdx.x * (size_t)blockDim.x + threadIdx.x) >> 5);
    const int lane = threadIdx.x & 31;
    if (wid >= ne) return;
    const int s = src[wid], d = dst[wid];
    for (int hh = 0; hh < heads; ++hh) {
        const float alpha = esc[wid * heads + hh] / denom[d * heads + hh];
        const float4 v = *(const float4*)(h + ((size_t)s * heads + hh) * GAT_F + lane * 4);
        float* o = rst + ((size_t)d * heads + hh) * GAT_F + lane * 4;
        atomicAdd(o + 0, alpha * v.x);
        atomicAdd(o + 1, alpha * v.y);
        atomicAdd(o + 2, alpha * v.z);
        atomicAdd(o + 3, alpha * v.w);
    }
}

// ---------------------------------------------------------------------------
extern "C" void kernel_launch(void* const* d_in, const int* in_sizes, int n_in,
                              void* d_out, int out_size, void* d_ws, size_t ws_size,
                              hipStream_t stream) {
    const float* feats   = (const float*)d_in[0];   // [N,128]
    const float* W1      = (const float*)d_in[1];   // [128,384]
    const float* attn_l1 = (const float*)d_in[2];   // [3,128]
    const float* attn_r1 = (const float*)d_in[3];   // [3,128]
    const float* bias1   = (const float*)d_in[4];   // [384]
    const float* W2      = (const float*)d_in[5];   // [384,128]
    const float* attn_l2 = (const float*)d_in[6];   // [1,128]
    const float* attn_r2 = (const float*)d_in[7];   // [1,128]
    const float* bias2   = (const float*)d_in[8];   // [128]
    const int*   src     = (const int*)d_in[9];     // [E]
    const int*   dst     = (const int*)d_in[10];    // [E]
    float* out = (float*)d_out;                     // [N,128]

    const int N = GAT_N, E = GAT_E, H = 3, HD = 3 * GAT_F;
    const int MT = N / 16;                          // 1250 row tiles

    // ---- workspace layout: f32 region first (all chunks 32B-multiples) ----
    float* ws     = (float*)d_ws;
    float* h1     = ws;                         // N*384
    float* rst1   = h1     + (size_t)N * HD;    // N*384
    float* el1    = rst1   + (size_t)N * HD;    // N*3
    float* er1    = el1    + (size_t)N * H;
    float* emax1  = er1    + (size_t)N * H;
    float* denom1 = emax1  + (size_t)N * H;
    float* esc1   = denom1 + (size_t)N * H;     // E*3
    float* h2     = esc1   + (size_t)E * H;     // N*128
    float* el2    = h2     + (size_t)N * GAT_F;
    float* er2    = el2    + (size_t)N;
    float* emax2  = er2    + (size_t)N;
    float* denom2 = emax2  + (size_t)N;
    float* esc2   = denom2 + (size_t)N;         // E
    // ---- f16 fragment region ----
    _Float16* Ap1 = (_Float16*)(esc2 + (size_t)E);   // N*128 halves
    _Float16* Bp1 = Ap1 + (size_t)N * GAT_F;         // 128*384 halves
    _Float16* Ap2 = Bp1 + (size_t)GAT_F * HD;        // N*384 halves
    _Float16* Bp2 = Ap2 + (size_t)N * HD;            // 384*128 halves

    const int TB = 256;

    // ===================== Layer 1 (H=3) =====================
    pack_a_frag<<<dim3(MT, GAT_F / 32), 32, 0, stream>>>(feats, Ap1, GAT_F, GAT_F / 32);
    pack_b_frag<<<dim3(HD / 16, GAT_F / 32), 32, 0, stream>>>(W1, Bp1, HD, GAT_F / 32);
    gemm_wmma_packed<3><<<MT, 256, 0, stream>>>(Ap1, Bp1, h1, GAT_F / 32, HD);

    attn_dots<<<(N * H * 32 + TB - 1) / TB, TB, 0, stream>>>(h1, attn_l1, attn_r1,
                                                             el1, er1, N, H);
    fill_f32<<<(N * H + TB - 1) / TB, TB, 0, stream>>>(emax1, -3.0e38f, N * H);
    fill_f32<<<(N * H + TB - 1) / TB, TB, 0, stream>>>(denom1, 0.f, N * H);
    init_bias<<<(N * HD + TB - 1) / TB, TB, 0, stream>>>(rst1, bias1, N, HD);
    edge_score<<<(E * H + TB - 1) / TB, TB, 0, stream>>>(src, dst, el1, er1, esc1,
                                                         emax1, E, H);
    edge_exp<<<(E * H + TB - 1) / TB, TB, 0, stream>>>(dst, esc1, emax1, denom1, E, H);
    edge_aggregate<<<(E * 32 + TB - 1) / TB, TB, 0, stream>>>(src, dst, esc1, denom1,
                                                              h1, rst1, E, H);

    // ===================== Layer 2 (H=1) =====================
    pack_a_frag<<<dim3(MT, HD / 32), 32, 0, stream>>>(rst1, Ap2, HD, HD / 32);
    pack_b_frag<<<dim3(GAT_F / 16, HD / 32), 32, 0, stream>>>(W2, Bp2, GAT_F, HD / 32);
    gemm_wmma_packed<1><<<MT, 256, 0, stream>>>(Ap2, Bp2, h2, HD / 32, GAT_F);

    attn_dots<<<(N * 32 + TB - 1) / TB, TB, 0, stream>>>(h2, attn_l2, attn_r2,
                                                         el2, er2, N, 1);
    fill_f32<<<(N + TB - 1) / TB, TB, 0, stream>>>(emax2, -3.0e38f, N);
    fill_f32<<<(N + TB - 1) / TB, TB, 0, stream>>>(denom2, 0.f, N);
    init_bias<<<(N * GAT_F + TB - 1) / TB, TB, 0, stream>>>(out, bias2, N, GAT_F);
    edge_score<<<(E + TB - 1) / TB, TB, 0, stream>>>(src, dst, el2, er2, esc2,
                                                     emax2, E, 1);
    edge_exp<<<(E + TB - 1) / TB, TB, 0, stream>>>(dst, esc2, emax2, denom2, E, 1);
    edge_aggregate<<<(E * 32 + TB - 1) / TB, TB, 0, stream>>>(src, dst, esc2, denom2,
                                                              h2, out, E, 1);
}
